// MultiHeadAttention_87101936763702
// MI455X (gfx1250) — matrix operand order repaired and checked by reading, not compiled, not run on hardware
//
#include <hip/hip_runtime.h>
#include <hip/hip_bf16.h>

// Problem constants (from reference)
constexpr int kB = 2, kS = 2048, kE = 1024, kH = 16, kHD = 64;
constexpr float kNEG = -1.0e24f;

typedef __attribute__((ext_vector_type(16))) _Float16 v16h;
typedef __attribute__((ext_vector_type(8)))  float    v8f;

#define WMMA_F16(a, b, c) \
  __builtin_amdgcn_wmma_f32_16x16x32_f16(false, (a), false, (b), (short)0, (c), false, false)

// ---------------------------------------------------------------------------
// fp32 -> fp16 elementwise convert
// ---------------------------------------------------------------------------
__global__ void cvt_f32_f16_kernel(const float* __restrict__ x,
                                   _Float16* __restrict__ y, int n) {
  int i = blockIdx.x * blockDim.x + threadIdx.x;
  if (i < n) y[i] = (_Float16)x[i];
}

// [H][E][HD] fp32 -> [E][H*HD] fp16 (flatten per-head weights into one GEMM)
__global__ void cvt_headw_kernel(const float* __restrict__ w,
                                 _Float16* __restrict__ y) {
  int i = blockIdx.x * blockDim.x + threadIdx.x;  // over H*E*HD
  if (i < kH * kE * kHD) {
    int hd = i % kHD;
    int e  = (i / kHD) % kE;
    int h  = i / (kHD * kE);
    y[(size_t)e * (kH * kHD) + h * kHD + hd] = (_Float16)w[i];
  }
}

// ---------------------------------------------------------------------------
// WMMA GEMM: C[M,N] = A[M,K](f16) * W[K,N](f16) + bias[N](f32)
// block = 256 threads (8 waves), tile 128x128, K-step 32.
// Wave w owns rows 16w..16w+15 and all 128 columns (8 accumulators).
// ---------------------------------------------------------------------------
template <bool F32OUT>
__global__ void __launch_bounds__(256)
gemm_wmma_kernel(const _Float16* __restrict__ A, const _Float16* __restrict__ W,
                 const float* __restrict__ bias, void* __restrict__ Cp,
                 int M, int N, int K) {
  __shared__ _Float16 Ash[128][40];   // [row][k], padded
  __shared__ _Float16 Bsh[128][40];   // [n][k] (transposed W tile), padded

  const int tid  = threadIdx.x;
  const int lane = tid & 31;
  const int wave = tid >> 5;          // 0..7
  const int ln   = lane & 15;
  const int hi   = lane >> 4;         // 0: lanes 0-15, 1: lanes 16-31

  const int r0 = blockIdx.y * 128;
  const int c0 = blockIdx.x * 128;

  v8f acc[8] = {v8f{}, v8f{}, v8f{}, v8f{}, v8f{}, v8f{}, v8f{}, v8f{}};

  const int kb = hi ? 8 : 0;          // unused; A frag uses db
  const int db = hi ? 16 : 0;         // B-frag K base
  const int arow = wave * 16 + ln;

  for (int k0 = 0; k0 < K; k0 += 32) {
    // Prefetch next K-step tiles into cache (gfx1250 global_prefetch_b8)
    if (k0 + 32 < K) {
      __builtin_prefetch(A + (size_t)(r0 + (tid & 127)) * K + k0 + 32, 0, 1);
      __builtin_prefetch(W + (size_t)(k0 + 32 + (tid & 31)) * N + c0 + ((tid >> 5) & 1) * 64, 0, 1);
    }
    __syncthreads();
    // Stage A tile: 128 rows x 32 halves (8-half segments; 512 segs / 256 thr)
    for (int idx = tid; idx < 128 * 4; idx += 256) {
      int row = idx >> 2, seg = (idx & 3) * 8;
      const _Float16* src = A + (size_t)(r0 + row) * K + k0 + seg;
#pragma unroll
      for (int i = 0; i < 8; ++i) Ash[row][seg + i] = src[i];
    }
    // Stage W tile transposed: Bsh[n][k] = W[k0+k][c0+n] (512 segs / 256 thr)
    for (int idx = tid; idx < 32 * 16; idx += 256) {
      int k = idx >> 4, nseg = (idx & 15) * 8;
      const _Float16* src = W + (size_t)(k0 + k) * N + c0 + nseg;
#pragma unroll
      for (int i = 0; i < 8; ++i) Bsh[nseg + i][k] = src[i];
    }
    __syncthreads();

    // A fragment (contiguous K: lo lanes k 0..15, hi lanes 16..31)
    v16h a;
#pragma unroll
    for (int i = 0; i < 8; ++i) {
      a[i]     = Ash[arow][db + i];
      a[i + 8] = Ash[arow][db + 8 + i];
    }
#pragma unroll
    for (int c = 0; c < 8; ++c) {
      v16h b;
      int n = c * 16 + ln;
#pragma unroll
      for (int i = 0; i < 16; ++i) b[i] = Bsh[n][db + i];
      acc[c] = WMMA_F16(a, b, acc[c]);
    }
  }

  // Epilogue: add bias, store (C layout: lane holds column n, rows r+8*hi)
#pragma unroll
  for (int c = 0; c < 8; ++c) {
    int n = c0 + c * 16 + ln;
    float bv = bias ? bias[n] : 0.0f;
#pragma unroll
    for (int r = 0; r < 8; ++r) {
      int row = r0 + wave * 16 + r + 8 * hi;
      float val = acc[c][r] + bv;
      if (F32OUT)
        ((float*)Cp)[(size_t)row * N + n] = val;
      else
        ((_Float16*)Cp)[(size_t)row * N + n] = (_Float16)val;
    }
  }
}

// ---------------------------------------------------------------------------
// Flash attention: per block = (b, h, 64-row q chunk); 4 waves x 16 q rows.
// 32 keys per iteration: two 16x16 score tiles (4 WMMAs over d), online
// softmax, then one full-K=32 P*V pass (4 WMMAs) — no zero padding.
// Q*K^T needs no transpose (Q and K are both d-major per lane per ISA §7.12.2);
// V is staged transposed in LDS for the P*V B-operand.
// ---------------------------------------------------------------------------
__global__ void __launch_bounds__(128)
flash_attn_kernel(const _Float16* __restrict__ qh, const _Float16* __restrict__ kh,
                  const _Float16* __restrict__ vh, const unsigned char* __restrict__ mask,
                  _Float16* __restrict__ out) {
  __shared__ _Float16 Ktile[32][64];      // [t][d]
  __shared__ _Float16 Vt[64][34];         // [d][t] (transposed, padded)
  __shared__ _Float16 Psh[4][16][32];     // per-wave prob tile [m][t]

  const int tid  = threadIdx.x;
  const int lane = tid & 31;
  const int wave = tid >> 5;
  const int ln   = lane & 15;
  const int hi   = lane >> 4;

  const int nqc = kS / 64;
  const int qc  = blockIdx.x % nqc;
  const int h   = (blockIdx.x / nqc) % kH;
  const int b   = blockIdx.x / (nqc * kH);
  const int s0  = qc * 64;

  const int kb = hi ? 8 : 0;
  const int db = hi ? 16 : 0;

  // Q fragments for both K-steps over d (HD=64 => d 0..31 and 32..63)
  const _Float16* qrow =
      qh + ((size_t)b * kS + s0 + wave * 16 + ln) * kE + h * kHD;
  v16h aq0, aq1;
#pragma unroll
  for (int i = 0; i < 8; ++i) {
    aq0[i]     = qrow[db + i];
    aq0[i + 8] = qrow[db + 8 + i];
    aq1[i]     = qrow[32 + db + i];
    aq1[i + 8] = qrow[32 + db + 8 + i];
  }

  // Online softmax state: this lane owns rows m = r + 8*hi of its wave's tile
  float mrun[8], lrun[8];
#pragma unroll
  for (int r = 0; r < 8; ++r) { mrun[r] = -1.0e30f; lrun[r] = 0.0f; }
  v8f acc[4] = {v8f{}, v8f{}, v8f{}, v8f{}};

  const unsigned char* mbase =
      mask + ((size_t)b * kS + s0 + wave * 16) * kS;

  for (int t0 = 0; t0 < kS; t0 += 32) {
    __syncthreads();
    // Stage K tile [t][d]: 32 rows x 8 segs = 256 segs / 128 threads
    for (int idx = tid; idx < 256; idx += 128) {
      int j = idx >> 3, dseg = (idx & 7) * 8;
      const _Float16* src = kh + ((size_t)b * kS + t0 + j) * kE + h * kHD + dseg;
#pragma unroll
      for (int i = 0; i < 8; ++i) Ktile[j][dseg + i] = src[i];
    }
    // Stage V tile transposed [d][t]
    for (int idx = tid; idx < 256; idx += 128) {
      int j = idx >> 3, dseg = (idx & 7) * 8;
      const _Float16* src = vh + ((size_t)b * kS + t0 + j) * kE + h * kHD + dseg;
#pragma unroll
      for (int i = 0; i < 8; ++i) Vt[dseg + i][j] = src[i];
    }
    __syncthreads();

    // scores for two 16-key tiles (B-frag: lane = column t, K = d)
    v16h bk00, bk01, bk10, bk11;
#pragma unroll
    for (int i = 0; i < 16; ++i) {
      bk00[i] = Ktile[ln][db + i];
      bk01[i] = Ktile[ln][32 + db + i];
      bk10[i] = Ktile[16 + ln][db + i];
      bk11[i] = Ktile[16 + ln][32 + db + i];
    }
    v8f s0v = {}, s1v = {};
    s0v = WMMA_F16(aq0, bk00, s0v);
    s0v = WMMA_F16(aq1, bk01, s0v);
    s1v = WMMA_F16(aq0, bk10, s1v);
    s1v = WMMA_F16(aq1, bk11, s1v);

    // scale + mask (tile0 col = t0+ln, tile1 col = t0+16+ln; row m = r+8*hi)
#pragma unroll
    for (int r = 0; r < 8; ++r) {
      int m = r + 8 * hi;
      float a0 = s0v[r] * 0.125f;  // 1/sqrt(64)
      float a1 = s1v[r] * 0.125f;
      if (mbase[(size_t)m * kS + t0 + ln]) a0 = kNEG;
      if (mbase[(size_t)m * kS + t0 + 16 + ln]) a1 = kNEG;
      s0v[r] = a0;
      s1v[r] = a1;
    }

    // online softmax: fold both tiles into one 16-lane shfl reduction tree
    float alpha[8];
#pragma unroll
    for (int r = 0; r < 8; ++r) {
      float rmax = fmaxf(s0v[r], s1v[r]);
#pragma unroll
      for (int off = 1; off < 16; off <<= 1)
        rmax = fmaxf(rmax, __shfl_xor(rmax, off, 32));
      float mnew = fmaxf(mrun[r], rmax);
      float a    = __expf(mrun[r] - mnew);
      float p0   = __expf(s0v[r] - mnew);
      float p1   = __expf(s1v[r] - mnew);
      float psum = p0 + p1;
#pragma unroll
      for (int off = 1; off < 16; off <<= 1)
        psum += __shfl_xor(psum, off, 32);
      lrun[r] = lrun[r] * a + psum;
      mrun[r] = mnew;
      alpha[r] = a;
      s0v[r] = p0;
      s1v[r] = p1;
    }
    // rescale output accumulators
#pragma unroll
    for (int c = 0; c < 4; ++c)
#pragma unroll
      for (int r = 0; r < 8; ++r) acc[c][r] *= alpha[r];

    // C-layout -> A-layout transpose of P through LDS (16x32 tile)
#pragma unroll
    for (int r = 0; r < 8; ++r) {
      Psh[wave][r + 8 * hi][ln]      = (_Float16)s0v[r];
      Psh[wave][r + 8 * hi][16 + ln] = (_Float16)s1v[r];
    }
    __syncthreads();

    // P fragment, full K=32 (A 16x32 layout: lo lanes K 0..15, hi 16..31)
    v16h ap;
#pragma unroll
    for (int i = 0; i < 8; ++i) {
      ap[i]     = Psh[wave][ln][db + i];
      ap[i + 8] = Psh[wave][ln][db + 8 + i];
    }

    // acc += P * V (B = V[t][d] 32x16: lane = column d, K = t split lo/hi)
#pragma unroll
    for (int c = 0; c < 4; ++c) {
      v16h bv;
#pragma unroll
      for (int i = 0; i < 16; ++i) bv[i] = Vt[c * 16 + ln][db + i];
      acc[c] = WMMA_F16(ap, bv, acc[c]);
    }
  }

  // finalize: divide by row sums, store concat-head layout [B*S][H*HD]
  float recip[8];
#pragma unroll
  for (int r = 0; r < 8; ++r) recip[r] = 1.0f / lrun[r];
#pragma unroll
  for (int c = 0; c < 4; ++c)
#pragma unroll
    for (int r = 0; r < 8; ++r) {
      int m = r + 8 * hi;
      out[((size_t)b * kS + s0 + wave * 16 + m) * kE + h * kHD + c * 16 + ln] =
          (_Float16)(acc[c][r] * recip[r]);
    }
}

// ---------------------------------------------------------------------------
// Launch
// ---------------------------------------------------------------------------
extern "C" void kernel_launch(void* const* d_in, const int* in_sizes, int n_in,
                              void* d_out, int out_size, void* d_ws, size_t ws_size,
                              hipStream_t stream) {
  const float* q    = (const float*)d_in[0];
  const float* k    = (const float*)d_in[1];
  const float* v    = (const float*)d_in[2];
  const unsigned char* mask = (const unsigned char*)d_in[3];
  const float* Wq = (const float*)d_in[4];  const float* bq = (const float*)d_in[5];
  const float* Wk = (const float*)d_in[6];  const float* bk = (const float*)d_in[7];
  const float* Wv = (const float*)d_in[8];  const float* bv = (const float*)d_in[9];
  const float* Whq = (const float*)d_in[10]; const float* bhq = (const float*)d_in[11];
  const float* Whk = (const float*)d_in[12]; const float* bhk = (const float*)d_in[13];
  const float* Whv = (const float*)d_in[14]; const float* bhv = (const float*)d_in[15];
  const float* Wo  = (const float*)d_in[16]; const float* bo  = (const float*)d_in[17];

  const int M = kB * kS;      // 4096
  const int N = kE;           // 1024
  const int K = kE;           // 1024
  const size_t MAT = (size_t)M * kE;   // 4M elems
  const size_t WEI = (size_t)kE * kE;  // 1M elems

  _Float16* p = (_Float16*)d_ws;
  _Float16* q16  = p; p += MAT;
  _Float16* k16  = p; p += MAT;
  _Float16* v16  = p; p += MAT;
  _Float16* wq16 = p; p += WEI;
  _Float16* wk16 = p; p += WEI;
  _Float16* wv16 = p; p += WEI;
  _Float16* wo16 = p; p += WEI;
  _Float16* whq16 = p; p += WEI;
  _Float16* whk16 = p; p += WEI;
  _Float16* whv16 = p; p += WEI;
  _Float16* qg16 = p; p += MAT;
  _Float16* kg16 = p; p += MAT;
  _Float16* vg16 = p; p += MAT;
  _Float16* qh16 = p; p += MAT;
  _Float16* kh16 = p; p += MAT;
  _Float16* vh16 = p; p += MAT;
  _Float16* at16 = p; p += MAT;

  const int cvt_blk = 256;
  // input activations
  cvt_f32_f16_kernel<<<(int)((MAT + 255) / 256), cvt_blk, 0, stream>>>(q, q16, (int)MAT);
  cvt_f32_f16_kernel<<<(int)((MAT + 255) / 256), cvt_blk, 0, stream>>>(k, k16, (int)MAT);
  cvt_f32_f16_kernel<<<(int)((MAT + 255) / 256), cvt_blk, 0, stream>>>(v, v16, (int)MAT);
  // weights
  cvt_f32_f16_kernel<<<(int)((WEI + 255) / 256), cvt_blk, 0, stream>>>(Wq, wq16, (int)WEI);
  cvt_f32_f16_kernel<<<(int)((WEI + 255) / 256), cvt_blk, 0, stream>>>(Wk, wk16, (int)WEI);
  cvt_f32_f16_kernel<<<(int)((WEI + 255) / 256), cvt_blk, 0, stream>>>(Wv, wv16, (int)WEI);
  cvt_f32_f16_kernel<<<(int)((WEI + 255) / 256), cvt_blk, 0, stream>>>(Wo, wo16, (int)WEI);
  cvt_headw_kernel<<<(int)((WEI + 255) / 256), cvt_blk, 0, stream>>>(Whq, whq16);
  cvt_headw_kernel<<<(int)((WEI + 255) / 256), cvt_blk, 0, stream>>>(Whk, whk16);
  cvt_headw_kernel<<<(int)((WEI + 255) / 256), cvt_blk, 0, stream>>>(Whv, whv16);

  dim3 ggrid(N / 128, M / 128);
  // global projections
  gemm_wmma_kernel<false><<<ggrid, 256, 0, stream>>>(q16, wq16, bq, qg16, M, N, K);
  gemm_wmma_kernel<false><<<ggrid, 256, 0, stream>>>(k16, wk16, bk, kg16, M, N, K);
  gemm_wmma_kernel<false><<<ggrid, 256, 0, stream>>>(v16, wv16, bv, vg16, M, N, K);
  // per-head projections (flattened to one 1024x1024 GEMM; bias is [H*HD] flat)
  gemm_wmma_kernel<false><<<ggrid, 256, 0, stream>>>(qg16, whq16, bhq, qh16, M, N, K);
  gemm_wmma_kernel<false><<<ggrid, 256, 0, stream>>>(kg16, whk16, bhk, kh16, M, N, K);
  gemm_wmma_kernel<false><<<ggrid, 256, 0, stream>>>(vg16, whv16, bhv, vh16, M, N, K);
  // flash attention
  flash_attn_kernel<<<kB * kH * (kS / 64), 128, 0, stream>>>(qh16, kh16, vh16, mask, at16);
  // output projection -> fp32 d_out
  gemm_wmma_kernel<true><<<ggrid, 256, 0, stream>>>(at16, wo16, bo, (float*)d_out, M, N, K);
}
